// ClassConditionalVQVAE_24068996727435
// MI455X (gfx1250) — matrix operand order repaired and checked by reading, not compiled, not run on hardware
//
#include <hip/hip_runtime.h>
#include <math.h>

// ---------------------------------------------------------------------------
// VQ-VAE forward for MI455X (gfx1250, wave32, WMMA + async LDS DMA).
//   B=32768, IN_DIM=2000, H1=512, H2=256, LAT=32, K=1024, BETA=0.25
// All GEMM dims: M is a multiple of 128, K is a multiple of 8, N of 4.
// Weights are pre-converted once per call to f16 [N][K] (transposed), so GEMM
// B-tiles are contiguous copies -> GLOBAL_LOAD_ASYNC_TO_LDS_B128 (ASYNCcnt).
// ---------------------------------------------------------------------------

typedef __attribute__((ext_vector_type(16))) _Float16     v16h;
typedef __attribute__((ext_vector_type(8)))  float        v8f;
typedef __attribute__((ext_vector_type(8)))  unsigned int v8u;

#define M_BLK 128
#define N_BLK 128
#define K_BLK 32
// 40 halves = 80 B row stride: fragment quads 16B-aligned (ds_load_b128),
// banks 20*m+c mod 64 conflict-free across 16 lanes.
#define LDSA_STRIDE 40

#define VQ_K    1024
#define VQ_LAT  32

#if defined(__has_builtin)
#if __has_builtin(__builtin_amdgcn_global_load_async_to_lds_b128)
#define HAVE_ASYNC_LDS 1
#endif
#endif

static __device__ __forceinline__ unsigned pack_h2(float a, float b) {
  return __builtin_bit_cast(unsigned, __builtin_amdgcn_cvt_pkrtz(a, b));
}

// 16-byte global->LDS copy: async DMA if available (no VGPR round-trip).
// Builtin payload type is gcc-vector int4; params are (global, lds, off, cpol).
typedef int async_b128_t __attribute__((vector_size(16)));
typedef __attribute__((address_space(1))) async_b128_t* g_b128p;
typedef __attribute__((address_space(3))) async_b128_t* l_b128p;

static __device__ __forceinline__ void copy16_g2l(
    const _Float16* src, _Float16* dst) {
#ifdef HAVE_ASYNC_LDS
  async_b128_t* s = (async_b128_t*)(uintptr_t)src;  // strip const, retype
  async_b128_t* d = (async_b128_t*)dst;
  __builtin_amdgcn_global_load_async_to_lds_b128((g_b128p)s, (l_b128p)d, 0, 0);
#else
  *(uint4*)dst = *(const uint4*)src;
#endif
}

// Wait for this wave's outstanding async LDS copies (before barrier).
static __device__ __forceinline__ void wait_async() {
#ifdef HAVE_ASYNC_LDS
#if __has_builtin(__builtin_amdgcn_s_wait_asynccnt)
  __builtin_amdgcn_s_wait_asynccnt(0);
#else
  asm volatile("s_wait_asynccnt 0x0" ::: "memory");
#endif
#endif
}

// Load a 16-half fragment as two 16B LDS reads (ds_load_b128).
static __device__ __forceinline__ v16h load_frag16(const _Float16* p0,
                                                   const _Float16* p1) {
  uint4 a = *(const uint4*)p0;
  uint4 b = *(const uint4*)p1;
  v8u t;
  t[0] = a.x; t[1] = a.y; t[2] = a.z; t[3] = a.w;
  t[4] = b.x; t[5] = b.y; t[6] = b.z; t[7] = b.w;
  return __builtin_bit_cast(v16h, t);
}

// ---------------------------------------------------------------------------
// Stage one 128x32 A tile (f32->f16 via VGPRs, loads batched first) and one
// 128x32 B tile (f16 [N][K] weights -> async 16B LDS copies).
// GUARDK=true only for the final partial k-block; guards are branchless:
// A zero-fills its k-tail (neutralizing B's clamped garbage rows); B and
// out-of-range N columns use clamped addresses (never stored by epilogue).
// ---------------------------------------------------------------------------
template <bool GUARDK>
static __device__ __forceinline__ void stage_tiles(
    const float* __restrict__ A, const _Float16* __restrict__ Bt,
    _Float16* __restrict__ As, _Float16* __restrict__ Bs,
    int tid, int block_m, int block_n, int k0g, int N, int K)
{
  // ---- B tile: 128 rows x 32 halves = 512 16B segments, 2 per thread
#pragma unroll
  for (int i = 0; i < 2; ++i) {
    int flat = tid + i * 256;
    int n   = flat >> 2;
    int seg = flat & 3;
    int n_g = min(block_n + n, N - 1);
    int k_g = GUARDK ? min(k0g + seg * 8, K - 8) : (k0g + seg * 8);
    copy16_g2l(Bt + (size_t)n_g * K + k_g, Bs + n * LDSA_STRIDE + seg * 8);
  }
  // ---- A tile: 128 rows x 32 cols = 1024 float4, 4 per thread (batched)
  float4 av[4];
#pragma unroll
  for (int i = 0; i < 4; ++i) {
    int flat4 = tid + i * 256;
    int m  = flat4 >> 3;
    int k4 = (flat4 & 7) << 2;
    int gk = k0g + k4;
    int gkc = GUARDK ? min(gk, K - 4) : gk;
    av[i] = *(const float4*)(A + (size_t)(block_m + m) * K + gkc);
    if (GUARDK && (gk + 3 >= K)) av[i] = make_float4(0.f, 0.f, 0.f, 0.f);
  }
#pragma unroll
  for (int i = 0; i < 4; ++i) {
    int flat4 = tid + i * 256;
    int m  = flat4 >> 3;
    int k4 = (flat4 & 7) << 2;
    *(uint2*)(As + m * LDSA_STRIDE + k4) =
        make_uint2(pack_h2(av[i].x, av[i].y), pack_h2(av[i].z, av[i].w));
  }
}

// Build the two A-fragments for this wave (ISA 16-bit A 16x32 layout).
static __device__ __forceinline__ void build_afrags(
    const _Float16* __restrict__ As, v16h afrag[2],
    int wm, int l16, int half16)
{
#pragma unroll
  for (int mt = 0; mt < 2; ++mt) {
    const _Float16* p = As + (wm * 32 + mt * 16 + l16) * LDSA_STRIDE + half16 * 8;
    afrag[mt] = load_frag16(p, p + 16);
  }
}

// Build the four B-fragments for this wave (ISA 16-bit B 32x16 layout).
static __device__ __forceinline__ void build_bfrags(
    const _Float16* __restrict__ Bs, v16h bfrag[4],
    int wn, int l16, int half16)
{
#pragma unroll
  for (int nt = 0; nt < 4; ++nt) {
    const _Float16* p = Bs + (wn * 64 + nt * 16 + l16) * LDSA_STRIDE + half16 * 8;
    bfrag[nt] = load_frag16(p, p + 16);
  }
}

// One 32-wide k-step: fragments + 8 WMMAs accumulating into acc.
static __device__ __forceinline__ void compute_step(
    const _Float16* __restrict__ As, const _Float16* __restrict__ Bs,
    v8f acc[2][4], int wm, int wn, int l16, int half16)
{
  v16h afrag[2], bfrag[4];
  build_afrags(As, afrag, wm, l16, half16);
  build_bfrags(Bs, bfrag, wn, l16, half16);
#pragma unroll
  for (int mt = 0; mt < 2; ++mt)
#pragma unroll
    for (int nt = 0; nt < 4; ++nt)
      acc[mt][nt] = __builtin_amdgcn_wmma_f32_16x16x32_f16(
          false, afrag[mt], false, bfrag[nt],
          (short)0, acc[mt][nt], false, false);
}

// ---------------------------------------------------------------------------
// Generic GEMM: C[M,N] = act(A[M,K] @ Bt[N,K]^T + bias[N]), act = relu or id.
// Block: 256 threads (8 waves), tile 128x128, k-step 32, double-buffered LDS.
// Waves arranged 4x2; each wave computes 32x64 via 2x4 WMMA f32_16x16x32_f16.
// Assumes: M % 128 == 0, K % 8 == 0, N % 4 == 0.
// ---------------------------------------------------------------------------
__global__ __launch_bounds__(256) void gemm_bias_act(
    const float* __restrict__ A, const _Float16* __restrict__ Bt,
    const float* __restrict__ bias, float* __restrict__ C,
    int M, int N, int K, int relu)
{
  __shared__ __align__(16) _Float16 As[2][M_BLK * LDSA_STRIDE];
  __shared__ __align__(16) _Float16 Bs[2][N_BLK * LDSA_STRIDE];

  const int tid    = threadIdx.x;
  const int lane   = tid & 31;
  const int wave   = tid >> 5;
  const int wm     = wave >> 1;    // 0..3 : 32-row slab within block tile
  const int wn     = wave & 1;     // 0..1 : 64-col slab
  const int half16 = lane >> 4;    // 0/1 (ISA lane-half)
  const int l16    = lane & 15;

  const int block_m = blockIdx.y * M_BLK;
  const int block_n = blockIdx.x * N_BLK;

  v8f acc[2][4] = {};

  const int nfull = K >> 5;        // number of full 32-wide k-blocks (>= 1 here)
  const int ktail = K & 31;

  stage_tiles<false>(A, Bt, As[0], Bs[0], tid, block_m, block_n, 0, N, K);
  wait_async();
  __syncthreads();

  for (int kt = 0; kt < nfull; ++kt) {
    const int cur = kt & 1;
    const int nxt = cur ^ 1;
    // prefetch next tile into the other buffer (global/async issue early)
    if (kt + 1 < nfull) {
      stage_tiles<false>(A, Bt, As[nxt], Bs[nxt],
                         tid, block_m, block_n, (kt + 1) * K_BLK, N, K);
    } else if (ktail) {
      stage_tiles<true>(A, Bt, As[nxt], Bs[nxt],
                        tid, block_m, block_n, nfull * K_BLK, N, K);
    }
    compute_step(As[cur], Bs[cur], acc, wm, wn, l16, half16);
    wait_async();
    __syncthreads();
  }
  if (ktail) {
    compute_step(As[nfull & 1], Bs[nfull & 1], acc, wm, wn, l16, half16);
  }

  // ---- epilogue: bias + optional relu, per ISA C/D layout
#pragma unroll
  for (int mt = 0; mt < 2; ++mt) {
#pragma unroll
    for (int nt = 0; nt < 4; ++nt) {
      int m_base = block_m + wm * 32 + mt * 16 + half16 * 8;
      int n_g    = block_n + wn * 64 + nt * 16 + l16;
      if (n_g < N) {
        float bv = bias[n_g];
#pragma unroll
        for (int r = 0; r < 8; ++r) {
          float val = acc[mt][nt][r] + bv;
          if (relu) val = fmaxf(val, 0.f);
          C[(size_t)(m_base + r) * N + n_g] = val;
        }
      }
    }
  }
}

// ---------------------------------------------------------------------------
// Weight prep: Wt[n][k] = (f16) W[k][n]  (coalesced writes)
// ---------------------------------------------------------------------------
__global__ void convert_weight_f16t(const float* __restrict__ W,
                                    _Float16* __restrict__ Wt, int K, int N) {
  int i = blockIdx.x * 256 + threadIdx.x;
  if (i < K * N) {
    int n = i / K, k = i - n * K;
    Wt[i] = (_Float16)W[(size_t)k * N + n];
  }
}

// Codebook prep: Et = (f16) E (already [n][k] layout); e2[c] = ||E_c||^2
__global__ void convert_e_f16(const float* __restrict__ E,
                              _Float16* __restrict__ Et,
                              float* __restrict__ e2) {
  int i = blockIdx.x * 256 + threadIdx.x;
  if (i < VQ_K * VQ_LAT) Et[i] = (_Float16)E[i];
  int c = i;
  if (c < VQ_K) {
    float s = 0.f;
#pragma unroll
    for (int d = 0; d < VQ_LAT; ++d) {
      float v = E[(size_t)c * VQ_LAT + d];
      s = fmaf(v, v, s);
    }
    e2[c] = s;
  }
}

// ---------------------------------------------------------------------------
// Fused WMMA vector quantization.
//   argmin_c ||z - e_c||^2 == argmin_c (||e_c||^2 - 2 z.e_c)
// Block = 256 threads (8 waves) owns 128 z-rows. Scores z.E^T computed with
// WMMA over 8 double-buffered chunks of 128 codes (chunks async-DMA'd to
// LDS); argmin fused per chunk so the 32768x1024 distance matrix never hits
// memory. z_q / (z_q - z_e)^2 use exact f32 z_e.
// ---------------------------------------------------------------------------
static __device__ __forceinline__ void stage_e_chunk(
    const _Float16* __restrict__ Et, _Float16* __restrict__ Bs,
    int tid, int ch)
{
#pragma unroll
  for (int i = 0; i < 2; ++i) {
    int flat = tid + i * 256;
    int c = flat >> 2, seg = flat & 3;
    copy16_g2l(Et + (size_t)(ch * 128 + c) * VQ_LAT + seg * 8,
               Bs + c * LDSA_STRIDE + seg * 8);
  }
}

__global__ __launch_bounds__(256) void vq_quantize_wmma(
    const float* __restrict__ Ze, const _Float16* __restrict__ Et,
    const float* __restrict__ E, const float* __restrict__ e2,
    float* __restrict__ Zq, float* __restrict__ code_out,
    float* __restrict__ counts, float* __restrict__ vq_accum)
{
  __shared__ __align__(16) _Float16 Asl[M_BLK * LDSA_STRIDE];     // z tile
  __shared__ __align__(16) _Float16 Bsl[2][N_BLK * LDSA_STRIDE];  // code chunks
  __shared__ float e2l[VQ_K];
  __shared__ float bestv_l[M_BLK][2];
  __shared__ int   besti_l[M_BLK][2];
  __shared__ float redsum[256];

  const int tid    = threadIdx.x;
  const int lane   = tid & 31;
  const int wave   = tid >> 5;
  const int wm     = wave >> 1;
  const int wn     = wave & 1;
  const int half16 = lane >> 4;
  const int l16    = lane & 15;
  const int row0   = blockIdx.x * M_BLK;

  // ---- stage z tile (128x32, f32->f16), e2, and first code chunk
  stage_e_chunk(Et, Bsl[0], tid, 0);
#pragma unroll
  for (int i = 0; i < 4; ++i) {
    int flat4 = tid + i * 256;
    int m = flat4 >> 3, k4 = (flat4 & 7) << 2;
    float4 v = *(const float4*)(Ze + (size_t)(row0 + m) * VQ_LAT + k4);
    *(uint2*)(Asl + m * LDSA_STRIDE + k4) =
        make_uint2(pack_h2(v.x, v.y), pack_h2(v.z, v.w));
  }
  for (int i = tid; i < VQ_K; i += 256) e2l[i] = e2[i];
  wait_async();
  __syncthreads();

  v16h afrag[2];
  build_afrags(Asl, afrag, wm, l16, half16);

  float best[2][8];
  int   bidx[2][8];
#pragma unroll
  for (int mt = 0; mt < 2; ++mt)
#pragma unroll
    for (int r = 0; r < 8; ++r) { best[mt][r] = 3.0e38f; bidx[mt][r] = 0; }

  for (int ch = 0; ch < VQ_K / 128; ++ch) {
    const int cur = ch & 1;
    const int nxt = cur ^ 1;
    if (ch + 1 < VQ_K / 128) stage_e_chunk(Et, Bsl[nxt], tid, ch + 1);

    v16h bfrag[4];
    build_bfrags(Bsl[cur], bfrag, wn, l16, half16);
    v8f acc[2][4] = {};
#pragma unroll
    for (int mt = 0; mt < 2; ++mt)
#pragma unroll
      for (int nt = 0; nt < 4; ++nt)
        acc[mt][nt] = __builtin_amdgcn_wmma_f32_16x16x32_f16(
            false, afrag[mt], false, bfrag[nt],
            (short)0, acc[mt][nt], false, false);

    // fused argmin update: d = e2[n] - 2*score (||z||^2 is row-constant)
#pragma unroll
    for (int nt = 0; nt < 4; ++nt) {
      int n = ch * 128 + wn * 64 + nt * 16 + l16;
      float e2v = e2l[n];
#pragma unroll
      for (int mt = 0; mt < 2; ++mt)
#pragma unroll
        for (int r = 0; r < 8; ++r) {
          float d = fmaf(-2.f, acc[mt][nt][r], e2v);
          if (d < best[mt][r]) { best[mt][r] = d; bidx[mt][r] = n; }
        }
    }
    wait_async();
    __syncthreads();
  }

  // ---- cross-lane argmin within each 16-lane half (rows live per half)
#pragma unroll
  for (int mt = 0; mt < 2; ++mt)
#pragma unroll
    for (int r = 0; r < 8; ++r) {
      float b = best[mt][r];
      int   ix = bidx[mt][r];
      for (int off = 1; off < 16; off <<= 1) {
        float ob = __shfl_xor(b, off, 32);
        int   oi = __shfl_xor(ix, off, 32);
        if (ob < b || (ob == b && oi < ix)) { b = ob; ix = oi; }
      }
      if (l16 == 0) {
        int row_local = wm * 32 + mt * 16 + half16 * 8 + r;
        bestv_l[row_local][wn] = b;
        besti_l[row_local][wn] = ix;
      }
    }
  __syncthreads();

  // ---- merge the two wn column-halves; emit code_idx + counts
  if (tid < M_BLK) {
    float b0 = bestv_l[tid][0], b1 = bestv_l[tid][1];
    int   i0 = besti_l[tid][0], i1 = besti_l[tid][1];
    int ix = (b1 < b0 || (b1 == b0 && i1 < i0)) ? i1 : i0;
    besti_l[tid][0] = ix;
    code_out[row0 + tid] = (float)ix;
    atomicAdd(&counts[ix], 1.0f);
  }
  __syncthreads();

  // ---- cooperative z_q gather + exact-f32 (z_q - z_e)^2 block reduction
  float lsum = 0.f;
  for (int i = tid; i < M_BLK * VQ_LAT; i += 256) {
    int rl = i >> 5, d = i & 31;
    int ix = besti_l[rl][0];
    float e = E[(size_t)ix * VQ_LAT + d];
    float z = Ze[(size_t)(row0 + rl) * VQ_LAT + d];
    float df = e - z;
    Zq[(size_t)(row0 + rl) * VQ_LAT + d] = e;
    lsum = fmaf(df, df, lsum);
  }
  redsum[tid] = lsum;
  __syncthreads();
  for (int s = 128; s > 0; s >>= 1) {
    if (tid < s) redsum[tid] += redsum[tid + s];
    __syncthreads();
  }
  if (tid == 0) atomicAdd(vq_accum, redsum[0]);
}

__global__ void vq_init(float* counts, float* accum) {
  int i = threadIdx.x + blockIdx.x * blockDim.x;
  if (i < VQ_K) counts[i] = 0.f;
  if (i == 0)   accum[0]  = 0.f;
}

__global__ __launch_bounds__(256) void vq_finalize(
    const float* __restrict__ counts, const float* __restrict__ accum,
    float* __restrict__ out_vq, float* __restrict__ out_usage)
{
  __shared__ float r1[256], r2[256];
  int tid = threadIdx.x;
  float tot = 0.f, nv = 0.f;
  for (int i = tid; i < VQ_K; i += 256) {
    float c = counts[i];
    tot += c;
    if (c > 0.f) nv += 1.f;
  }
  r1[tid] = tot; r2[tid] = nv;
  __syncthreads();
  for (int s = 128; s > 0; s >>= 1) {
    if (tid < s) { r1[tid] += r1[tid + s]; r2[tid] += r2[tid + s]; }
    __syncthreads();
  }
  float total = r1[0] + 1e-8f;
  float nvalid = r2[0];
  __syncthreads();
  float u = 0.f;
  for (int i = tid; i < VQ_K; i += 256) {
    float c = counts[i];
    if (c > 0.f) { float p = c / total; u += p * logf(p * nvalid); }
  }
  r1[tid] = u;
  __syncthreads();
  for (int s = 128; s > 0; s >>= 1) {
    if (tid < s) r1[tid] += r1[tid + s];
    __syncthreads();
  }
  if (tid == 0) {
    *out_usage = r1[0];
    *out_vq = accum[0] * 1.25f / (32768.f * 32.f);  // (1 + BETA) * mean
  }
}

// ---------------------------------------------------------------------------
extern "C" void kernel_launch(void* const* d_in, const int* in_sizes, int n_in,
                              void* d_out, int out_size, void* d_ws, size_t ws_size,
                              hipStream_t stream) {
  (void)in_sizes; (void)n_in; (void)out_size; (void)ws_size;

  const int B = 32768, IN_DIM = 2000, H1 = 512, H2 = 256, LAT = 32;

  const float* x   = (const float*)d_in[0];
  // d_in[1] = y (unused by the reference computation)
  const float* We1 = (const float*)d_in[2];
  const float* be1 = (const float*)d_in[3];
  const float* We2 = (const float*)d_in[4];
  const float* be2 = (const float*)d_in[5];
  const float* We3 = (const float*)d_in[6];
  const float* be3 = (const float*)d_in[7];
  const float* Wd1 = (const float*)d_in[8];
  const float* bd1 = (const float*)d_in[9];
  const float* Wd2 = (const float*)d_in[10];
  const float* bd2 = (const float*)d_in[11];
  const float* Wd3 = (const float*)d_in[12];
  const float* bd3 = (const float*)d_in[13];
  const float* E   = (const float*)d_in[14];

  // workspace layout (16B alignment maintained; accum last)
  float* buf1   = (float*)d_ws;                       // B x 512 (H1 / dec-H1)
  float* buf2   = buf1 + (size_t)B * H1;              // B x 256 (H2 / dec-H2)
  float* Ze     = buf2 + (size_t)B * H2;              // B x 32
  float* Zq     = Ze   + (size_t)B * LAT;             // B x 32
  float* counts = Zq   + (size_t)B * LAT;             // 1024
  float* e2     = counts + VQ_K;                      // 1024
  _Float16* Wt_e1 = (_Float16*)(e2 + VQ_K);           // [512 ][2000]
  _Float16* Wt_e2 = Wt_e1 + (size_t)H1 * IN_DIM;      // [256 ][512 ]
  _Float16* Wt_e3 = Wt_e2 + (size_t)H2 * H1;          // [32  ][256 ]
  _Float16* Wt_d1 = Wt_e3 + (size_t)LAT * H2;         // [256 ][32  ]
  _Float16* Wt_d2 = Wt_d1 + (size_t)H2 * LAT;         // [512 ][256 ]
  _Float16* Wt_d3 = Wt_d2 + (size_t)H1 * H2;          // [2000][512 ]
  _Float16* Et    = Wt_d3 + (size_t)IN_DIM * H1;      // [1024][32  ]
  float* accum    = (float*)(Et + (size_t)VQ_K * VQ_LAT);

  // output layout: x_rec | vq_loss | usage_loss | code_idx
  float* x_rec     = (float*)d_out;
  float* out_vq    = x_rec + (size_t)B * IN_DIM;
  float* out_usage = out_vq + 1;
  float* code_out  = out_usage + 1;

  dim3 blk(256);
  vq_init<<<dim3((VQ_K + 255) / 256), blk, 0, stream>>>(counts, accum);
  convert_e_f16<<<dim3((VQ_K * VQ_LAT + 255) / 256), blk, 0, stream>>>(E, Et, e2);

  // ---- one-time weight prep: f16, transposed to [N][K]
#define CONV_W(W, WT, Kd, Nd)                                              \
  convert_weight_f16t<<<dim3(((Kd) * (Nd) + 255) / 256), blk, 0, stream>>>( \
      (W), (WT), (Kd), (Nd));
  CONV_W(We1, Wt_e1, IN_DIM, H1);
  CONV_W(We2, Wt_e2, H1, H2);
  CONV_W(We3, Wt_e3, H2, LAT);
  CONV_W(Wd1, Wt_d1, LAT, H2);
  CONV_W(Wd2, Wt_d2, H2, H1);
  CONV_W(Wd3, Wt_d3, H1, IN_DIM);
#undef CONV_W

  // ---- encoder
  gemm_bias_act<<<dim3(H1 / N_BLK, B / M_BLK), blk, 0, stream>>>(
      x, Wt_e1, be1, buf1, B, H1, IN_DIM, 1);
  gemm_bias_act<<<dim3(H2 / N_BLK, B / M_BLK), blk, 0, stream>>>(
      buf1, Wt_e2, be2, buf2, B, H2, H1, 1);
  gemm_bias_act<<<dim3(1, B / M_BLK), blk, 0, stream>>>(
      buf2, Wt_e3, be3, Ze, B, LAT, H2, 0);

  // ---- vector quantization (WMMA-fused argmin)
  vq_quantize_wmma<<<dim3(B / M_BLK), blk, 0, stream>>>(
      Ze, Et, E, e2, Zq, code_out, counts, accum);
  vq_finalize<<<dim3(1), blk, 0, stream>>>(counts, accum, out_vq, out_usage);

  // ---- decoder
  gemm_bias_act<<<dim3(H2 / N_BLK, B / M_BLK), blk, 0, stream>>>(
      Zq, Wt_d1, bd1, buf2, B, H2, LAT, 1);
  gemm_bias_act<<<dim3(H1 / N_BLK, B / M_BLK), blk, 0, stream>>>(
      buf2, Wt_d2, bd2, buf1, B, H1, H2, 1);
  gemm_bias_act<<<dim3((IN_DIM + N_BLK - 1) / N_BLK, B / M_BLK), blk, 0, stream>>>(
      buf1, Wt_d3, bd3, x_rec, B, IN_DIM, H1, 0);
}
